// EnsembleModel_86363202387983
// MI455X (gfx1250) — matrix operand order repaired and checked by reading, not compiled, not run on hardware
//
#include <hip/hip_runtime.h>
#include <hip/hip_bf16.h>

typedef _Float16 half_t;
typedef _Float16 v16h __attribute__((ext_vector_type(16)));
typedef _Float16 v8h  __attribute__((ext_vector_type(8)));
typedef float    v8f  __attribute__((ext_vector_type(8)));

#define KEXP 4
#define PNUM 4000
#define EDIM 128
#define HDIM 256
#define NSEQ 2048   // NB*B
#define TSEQ 64
#define NTOK 16000  // KEXP*PNUM

static __device__ __forceinline__ float sigmoidf_(float x) { return 1.0f / (1.0f + __expf(-x)); }

// A fragment (16xK tile, row-major, fixed row per lane):
// lane<16: halves[0..7]=K kb..kb+7, halves[8..15]=K kb+16..kb+23, kb=0 ; lane>=16: kb=8
static __device__ __forceinline__ v16h load_a(const half_t* base, int ld, int row, int koff, int lane) {
    int kb = (lane < 16) ? 0 : 8;
    const half_t* p = base + (size_t)row * ld + koff + kb;
    v8h lo = *(const v8h*)(p);
    v8h hh = *(const v8h*)(p + 16);
    v16h r;
#pragma unroll
    for (int i = 0; i < 8; ++i) { r[i] = lo[i]; r[i + 8] = hh[i]; }
    return r;
}

// B fragment (Kx16 tile; per lane: fixed column):
// lane<16: halves[0..7]=K kb..kb+7, halves[8..15]=K kb+16..kb+23, kb=0 ; lane>=16: kb=8
static __device__ __forceinline__ v16h load_b(const half_t* base, int ld, int col, int koff, int lane) {
    int kb = (lane < 16) ? 0 : 8;
    const half_t* p = base + (size_t)col * ld + koff + kb;
    v8h lo = *(const v8h*)(p);
    v8h hh = *(const v8h*)(p + 16);
    v16h r;
#pragma unroll
    for (int i = 0; i < 8; ++i) { r[i] = lo[i]; r[i + 8] = hh[i]; }
    return r;
}

#define WMMA_F16(a, b, c) __builtin_amdgcn_wmma_f32_16x16x32_f16(false, (a), false, (b), (short)0, (c), false, false)

// ---------------- Kernel 0: weight + embedding conversion fp32 -> f16, fused bias ----------------
#define N_WIH (KEXP * 1024 * 128)
#define N_WHH (KEXP * 1024 * 256)
#define N_W1  (KEXP * 256 * 256)
#define N_W2  (KEXP * 128 * 256)
#define N_WQ  (128 * 128)
#define N_BC  (KEXP * 1024)
#define N_EMB (KEXP * PNUM * EDIM)
#define N_CVT (N_WIH + N_WHH + N_W1 + N_W2 + 3 * N_WQ + N_BC + N_EMB)

__global__ __launch_bounds__(256) void convert_kernel(
    const float* Wih, const float* Whh, const float* W1, const float* W2,
    const float* Wq, const float* Wk, const float* Wv,
    const float* bih, const float* bhh, const float* emb,
    half_t* WihH, half_t* WhhH, half_t* W1H, half_t* W2H,
    half_t* WqH, half_t* WkH, half_t* WvH, float* biasc, half_t* embH) {
    size_t i = (size_t)blockIdx.x * 256 + threadIdx.x;
    if (i < N_WIH) { WihH[i] = (half_t)Wih[i]; return; } i -= N_WIH;
    if (i < N_WHH) { WhhH[i] = (half_t)Whh[i]; return; } i -= N_WHH;
    if (i < N_W1)  { W1H[i]  = (half_t)W1[i];  return; } i -= N_W1;
    if (i < N_W2)  { W2H[i]  = (half_t)W2[i];  return; } i -= N_W2;
    if (i < N_WQ)  { WqH[i]  = (half_t)Wq[i];  return; } i -= N_WQ;
    if (i < N_WQ)  { WkH[i]  = (half_t)Wk[i];  return; } i -= N_WQ;
    if (i < N_WQ)  { WvH[i]  = (half_t)Wv[i];  return; } i -= N_WQ;
    if (i < N_BC)  { biasc[i] = bih[i] + bhh[i]; return; } i -= N_BC;
    if (i < N_EMB) { embH[i] = (half_t)emb[i]; }
}

// ---------------- Kernel 1: embedding gather + LSTM(T=64) + MLP ----------------
// grid = (NSEQ/16, KEXP), block = 256 (8 waves). Each block: 16 rows.
// Wave wv owns hidden units [wv*32, wv*32+32) for all 4 gates -> cell state in regs.
__global__ __launch_bounds__(256) void lstm_kernel(
    const int* seqs, const half_t* embH,
    const half_t* WihH, const half_t* WhhH, const float* biasc,
    const half_t* W1H, const float* b1, const half_t* W2H, const float* b2,
    float* outs) {
    const int k = blockIdx.y;
    const int n0 = blockIdx.x * 16;
    const int tid = threadIdx.x;
    const int wv = tid >> 5;
    const int lane = tid & 31;
    const int lrow = lane & 15;
    const int crow = (lane < 16) ? 0 : 8;  // C/D layout row offset

    __shared__ __align__(16) half_t xA[16 * 128];
    __shared__ __align__(16) half_t hA[16 * 256];
    __shared__ __align__(16) half_t mA[16 * 256];

    for (int i = tid; i < 16 * 256; i += 256) hA[i] = (half_t)0.0f;

    // per-lane bias for owned gate columns
    float bgate[4][2];
#pragma unroll
    for (int g = 0; g < 4; ++g)
#pragma unroll
        for (int q = 0; q < 2; ++q)
            bgate[g][q] = biasc[(size_t)k * 1024 + g * 256 + wv * 32 + q * 16 + lrow];

    v8f czero = {0.f, 0.f, 0.f, 0.f, 0.f, 0.f, 0.f, 0.f};
    v8f cacc[2] = {czero, czero};

    const half_t* WihE = WihH + (size_t)k * 1024 * 128;
    const half_t* WhhE = WhhH + (size_t)k * 1024 * 256;

    // per-thread gather slot: row = tid>>4, 8-half segment seg = tid&15
    const int grow = tid >> 4, gseg = tid & 15;
    const int* seqp = seqs + ((size_t)k * NSEQ + n0 + grow) * TSEQ;

    __syncthreads();
    for (int t = 0; t < TSEQ; ++t) {
        // gather x_t tile [16][128] f16 -> LDS (single 16B load per thread)
        {
            int s = seqp[t];
            *(v8h*)(xA + grow * 128 + gseg * 8) =
                *(const v8h*)(embH + ((size_t)k * PNUM + s) * EDIM + gseg * 8);
        }
        __syncthreads();

        v8f acc[4][2];
#pragma unroll
        for (int g = 0; g < 4; ++g) { acc[g][0] = czero; acc[g][1] = czero; }

        // x_t @ Wih^T  (K = 128)
#pragma unroll
        for (int kk = 0; kk < 4; ++kk) {
            v16h a = load_a(xA, 128, lrow, kk * 32, lane);
#pragma unroll
            for (int g = 0; g < 4; ++g)
#pragma unroll
                for (int q = 0; q < 2; ++q) {
                    int j = g * 256 + wv * 32 + q * 16 + lrow;
                    v16h b = load_b(WihE, 128, j, kk * 32, lane);
                    acc[g][q] = WMMA_F16(a, b, acc[g][q]);
                }
        }
        // h @ Whh^T  (K = 256)
#pragma unroll
        for (int kk = 0; kk < 8; ++kk) {
            v16h a = load_a(hA, 256, lrow, kk * 32, lane);
#pragma unroll
            for (int g = 0; g < 4; ++g)
#pragma unroll
                for (int q = 0; q < 2; ++q) {
                    int j = g * 256 + wv * 32 + q * 16 + lrow;
                    v16h b = load_b(WhhE, 256, j, kk * 32, lane);
                    acc[g][q] = WMMA_F16(a, b, acc[g][q]);
                }
        }
        __syncthreads();  // all waves done reading hA

        // gate activations (torch order i,f,g,o); c stays in registers
#pragma unroll
        for (int q = 0; q < 2; ++q)
#pragma unroll
            for (int r = 0; r < 8; ++r) {
                float gi = acc[0][q][r] + bgate[0][q];
                float gf = acc[1][q][r] + bgate[1][q];
                float gg = acc[2][q][r] + bgate[2][q];
                float go = acc[3][q][r] + bgate[3][q];
                float c = sigmoidf_(gf) * cacc[q][r] + sigmoidf_(gi) * tanhf(gg);
                float h = sigmoidf_(go) * tanhf(c);
                cacc[q][r] = c;
                hA[(r + crow) * 256 + wv * 32 + q * 16 + lrow] = (half_t)h;
            }
        __syncthreads();
    }

    // m = LeakyReLU(h @ W1^T + b1)  [16,256]; wave covers cols [wv*32, wv*32+32)
    v8f macc[2] = {czero, czero};
#pragma unroll
    for (int kk = 0; kk < 8; ++kk) {
        v16h a = load_a(hA, 256, lrow, kk * 32, lane);
#pragma unroll
        for (int q = 0; q < 2; ++q) {
            int j = wv * 32 + q * 16 + lrow;
            v16h b = load_b(W1H + (size_t)k * 256 * 256, 256, j, kk * 32, lane);
            macc[q] = WMMA_F16(a, b, macc[q]);
        }
    }
    __syncthreads();
#pragma unroll
    for (int q = 0; q < 2; ++q)
#pragma unroll
        for (int r = 0; r < 8; ++r) {
            int col = wv * 32 + q * 16 + lrow;
            float v = macc[q][r] + b1[(size_t)k * 256 + col];
            v = (v >= 0.f) ? v : 0.2f * v;
            mA[(r + crow) * 256 + col] = (half_t)v;
        }
    __syncthreads();

    // out = m @ W2^T + b2  [16,128]; wave covers cols [wv*16, wv*16+16)
    v8f oacc = czero;
#pragma unroll
    for (int kk = 0; kk < 8; ++kk) {
        v16h a = load_a(mA, 256, lrow, kk * 32, lane);
        v16h b = load_b(W2H + (size_t)k * 128 * 256, 256, wv * 16 + lrow, kk * 32, lane);
        oacc = WMMA_F16(a, b, oacc);
    }
#pragma unroll
    for (int r = 0; r < 8; ++r) {
        int col = wv * 16 + lrow;
        outs[((size_t)k * NSEQ + n0 + r + crow) * EDIM + col] = oacc[r] + b2[(size_t)k * 128 + col];
    }
}

// ---------------- Kernel 2: zero, scatter-add, normalize ----------------
__global__ __launch_bounds__(256) void zero_kernel(float* eo, float* counts) {
    size_t i = (size_t)blockIdx.x * 256 + threadIdx.x;
    if (i < (size_t)KEXP * PNUM * EDIM) eo[i] = 0.0f;
    if (i < PNUM) counts[i] = 0.0f;
}

__global__ __launch_bounds__(128) void scatter_kernel(
    const float* outs, const int* pidx, float* eo, float* counts) {
    int bidx = blockIdx.x;              // K*NSEQ
    int k = bidx >> 11;
    int p = pidx[bidx];
    int e = threadIdx.x;                // 128
    atomicAdd(&eo[((size_t)k * PNUM + p) * EDIM + e], outs[(size_t)bidx * EDIM + e]);
    if (e == 0) atomicAdd(&counts[p], 1.0f);
}

__global__ __launch_bounds__(256) void norm_kernel(
    const float* eo, const float* counts, half_t* xfH) {
    size_t i = (size_t)blockIdx.x * 256 + threadIdx.x;
    if (i < (size_t)KEXP * PNUM * EDIM) {
        int p = (int)((i >> 7) % PNUM);
        xfH[i] = (half_t)(eo[i] / counts[p]);
    }
}

// ---------------- Kernel 3: Q/K/V projections ----------------
// grid = (NTOK/16, 3); Q,K row-major f16; V stored transposed [E][NTOK] f16.
__global__ __launch_bounds__(256) void qkv_kernel(
    const half_t* xfH, const half_t* WqH, const half_t* WkH, const half_t* WvH,
    const float* bq, const float* bk, const float* bv,
    half_t* Qh, half_t* Kh, half_t* VtH) {
    const int i0 = blockIdx.x * 16;
    const int which = blockIdx.y;
    const int tid = threadIdx.x, wv = tid >> 5, lane = tid & 31;
    const int lrow = lane & 15, crow = (lane < 16) ? 0 : 8;
    const half_t* W = (which == 0) ? WqH : (which == 1) ? WkH : WvH;
    const float* bias = (which == 0) ? bq : (which == 1) ? bk : bv;

    v8f acc = {0.f, 0.f, 0.f, 0.f, 0.f, 0.f, 0.f, 0.f};
#pragma unroll
    for (int kk = 0; kk < 4; ++kk) {
        v16h a = load_a(xfH, 128, i0 + lrow, kk * 32, lane);
        v16h b = load_b(W, 128, wv * 16 + lrow, kk * 32, lane);
        acc = WMMA_F16(a, b, acc);
    }
#pragma unroll
    for (int r = 0; r < 8; ++r) {
        int row = i0 + r + crow;
        int col = wv * 16 + lrow;
        half_t v = (half_t)(acc[r] + bias[col]);
        if (which == 0)      Qh[(size_t)row * 128 + col] = v;
        else if (which == 1) Kh[(size_t)row * 128 + col] = v;
        else                 VtH[(size_t)col * NTOK + row] = v;
    }
}

// ---------------- Kernel 4: flash attention over 16000 tokens ----------------
// grid = NTOK/16; 125 chunks of 128 keys; never materializes the 16000^2 matrix.
__global__ __launch_bounds__(256) void attn_kernel(
    const half_t* Qh, const half_t* Kh, const half_t* VtH, float* out) {
    const int i0 = blockIdx.x * 16;
    const int tid = threadIdx.x, wv = tid >> 5, lane = tid & 31;
    const int lrow = lane & 15, crow = (lane < 16) ? 0 : 8;

    __shared__ __align__(16) float  sT[16 * 128];
    __shared__ __align__(16) half_t pT[16 * 128];
    __shared__ float pmax[16 * 16];
    __shared__ float psum[16 * 16];
    __shared__ float mrun[16], lrun[16], fac[16];

    if (tid < 16) { mrun[tid] = -1e30f; lrun[tid] = 0.0f; }

    v16h qa[4];
#pragma unroll
    for (int kk = 0; kk < 4; ++kk) qa[kk] = load_a(Qh, 128, i0 + lrow, kk * 32, lane);

    v8f accO = {0.f, 0.f, 0.f, 0.f, 0.f, 0.f, 0.f, 0.f};
    const float sc = 0.08838834764831845f;  // 1/sqrt(128)
    __syncthreads();

    for (int jc = 0; jc < NTOK / 128; ++jc) {
        int j0 = jc * 128;
        // S tile [16][128]: wave wv computes cols [wv*16, wv*16+16)
        v8f s = {0.f, 0.f, 0.f, 0.f, 0.f, 0.f, 0.f, 0.f};
#pragma unroll
        for (int kk = 0; kk < 4; ++kk) {
            v16h b = load_b(Kh, 128, j0 + wv * 16 + lrow, kk * 32, lane);
            s = WMMA_F16(qa[kk], b, s);
        }
#pragma unroll
        for (int r = 0; r < 8; ++r) sT[(r + crow) * 128 + wv * 16 + lrow] = s[r] * sc;
        __syncthreads();

        // per-row chunk max (16 threads/row x 8 cols)
        {
            int r = tid >> 4, c = tid & 15;
            float mx = -1e30f;
#pragma unroll
            for (int i = 0; i < 8; ++i) mx = fmaxf(mx, sT[r * 128 + c * 8 + i]);
            pmax[r * 16 + c] = mx;
        }
        __syncthreads();
        if (tid < 16) {
            float mx = mrun[tid];
            for (int c = 0; c < 16; ++c) mx = fmaxf(mx, pmax[tid * 16 + c]);
            fac[tid] = __expf(mrun[tid] - mx);
            mrun[tid] = mx;
        }
        __syncthreads();
        // P = exp(S - m), partial sums
        {
            int r = tid >> 4, c = tid & 15;
            float m = mrun[r], ss = 0.0f;
#pragma unroll
            for (int i = 0; i < 8; ++i) {
                float p = __expf(sT[r * 128 + c * 8 + i] - m);
                pT[r * 128 + c * 8 + i] = (half_t)p;
                ss += p;
            }
            psum[r * 16 + c] = ss;
        }
        __syncthreads();
        if (tid < 16) {
            float ss = 0.0f;
            for (int c = 0; c < 16; ++c) ss += psum[tid * 16 + c];
            lrun[tid] = lrun[tid] * fac[tid] + ss;
        }
        __syncthreads();

        // rescale running accumulator, then accO += P @ Vchunk
#pragma unroll
        for (int r = 0; r < 8; ++r) accO[r] *= fac[r + crow];
#pragma unroll
        for (int kk = 0; kk < 4; ++kk) {
            v16h a = load_a(pT, 128, lrow, kk * 32, lane);
            v16h b = load_b(VtH, NTOK, wv * 16 + lrow, j0 + kk * 32, lane);
            accO = WMMA_F16(a, b, accO);
        }
        __syncthreads();
    }

#pragma unroll
    for (int r = 0; r < 8; ++r)
        out[(size_t)(i0 + r + crow) * EDIM + wv * 16 + lrow] = accO[r] / lrun[r + crow];
}

// ---------------- host ----------------
extern "C" void kernel_launch(void* const* d_in, const int* in_sizes, int n_in,
                              void* d_out, int out_size, void* d_ws, size_t ws_size,
                              hipStream_t stream) {
    const int*   seqs = (const int*)d_in[0];
    const int*   pidx = (const int*)d_in[1];
    const float* emb  = (const float*)d_in[2];
    const float* Wih  = (const float*)d_in[3];
    const float* Whh  = (const float*)d_in[4];
    const float* bih  = (const float*)d_in[5];
    const float* bhh  = (const float*)d_in[6];
    const float* W1   = (const float*)d_in[7];
    const float* b1   = (const float*)d_in[8];
    const float* W2   = (const float*)d_in[9];
    const float* b2   = (const float*)d_in[10];
    const float* Wq   = (const float*)d_in[11];
    const float* bq   = (const float*)d_in[12];
    const float* Wk   = (const float*)d_in[13];
    const float* bk   = (const float*)d_in[14];
    const float* Wv   = (const float*)d_in[15];
    const float* bv   = (const float*)d_in[16];

    char* ws = (char*)d_ws;
    size_t off = 0;
    auto alloc = [&](size_t bytes) -> void* {
        void* p = ws + off;
        off = (off + bytes + 255) & ~(size_t)255;
        return p;
    };
    half_t* WihH = (half_t*)alloc((size_t)N_WIH * 2);
    half_t* WhhH = (half_t*)alloc((size_t)N_WHH * 2);
    half_t* W1H  = (half_t*)alloc((size_t)N_W1 * 2);
    half_t* W2H  = (half_t*)alloc((size_t)N_W2 * 2);
    half_t* WqH  = (half_t*)alloc((size_t)N_WQ * 2);
    half_t* WkH  = (half_t*)alloc((size_t)N_WQ * 2);
    half_t* WvH  = (half_t*)alloc((size_t)N_WQ * 2);
    float*  biasc = (float*)alloc((size_t)N_BC * 4);
    half_t* embH = (half_t*)alloc((size_t)N_EMB * 2);
    float*  outs  = (float*)alloc((size_t)KEXP * NSEQ * EDIM * 4);
    float*  eo    = (float*)alloc((size_t)KEXP * PNUM * EDIM * 4);
    float*  counts = (float*)alloc((size_t)PNUM * 4);
    half_t* xfH  = (half_t*)alloc((size_t)NTOK * EDIM * 2);
    half_t* Qh   = (half_t*)alloc((size_t)NTOK * EDIM * 2);
    half_t* Kh   = (half_t*)alloc((size_t)NTOK * EDIM * 2);
    half_t* VtH  = (half_t*)alloc((size_t)EDIM * NTOK * 2);
    (void)ws_size; (void)in_sizes; (void)n_in; (void)out_size;

    convert_kernel<<<(N_CVT + 255) / 256, 256, 0, stream>>>(
        Wih, Whh, W1, W2, Wq, Wk, Wv, bih, bhh, emb,
        WihH, WhhH, W1H, W2H, WqH, WkH, WvH, biasc, embH);

    lstm_kernel<<<dim3(NSEQ / 16, KEXP), 256, 0, stream>>>(
        seqs, embH, WihH, WhhH, biasc, W1H, b1, W2H, b2, outs);

    zero_kernel<<<(KEXP * PNUM * EDIM + 255) / 256, 256, 0, stream>>>(eo, counts);

    scatter_kernel<<<KEXP * NSEQ, 128, 0, stream>>>(outs, pidx, eo, counts);

    norm_kernel<<<(KEXP * PNUM * EDIM + 255) / 256, 256, 0, stream>>>(eo, counts, xfH);

    qkv_kernel<<<dim3(NTOK / 16, 3), 256, 0, stream>>>(
        xfH, WqH, WkH, WvH, bq, bk, bv, Qh, Kh, VtH);

    attn_kernel<<<NTOK / 16, 256, 0, stream>>>(Qh, Kh, VtH, (float*)d_out);
}
